// LocalAttentionBlock_15848429322820
// MI455X (gfx1250) — compile-verified, matrix operand order checked by reading, not run on hardware
//
#include <hip/hip_runtime.h>
#include <math.h>

// ---------------------------------------------------------------------------
// Types / helpers
// ---------------------------------------------------------------------------
typedef __attribute__((ext_vector_type(16))) __bf16 v16bf;
typedef __attribute__((ext_vector_type(8)))  __bf16 v8bf;
typedef __attribute__((ext_vector_type(8)))  float  v8f;

// exact pointee type expected by the async-load builtin (per hipcc diagnostic)
typedef int v4i_t __attribute__((vector_size(4 * sizeof(int))));
typedef __attribute__((address_space(1))) v4i_t* glob128_t;
typedef __attribute__((address_space(3))) v4i_t* lds128_t;

#ifndef __has_builtin
#define __has_builtin(x) 0
#endif

#if defined(__AMDGCN__) && __has_builtin(__builtin_amdgcn_global_load_async_to_lds_b128)
#define HAVE_ASYNC_LDS 1
#endif

__device__ __forceinline__ void wait_async0() {
#if defined(__AMDGCN__)
#if __has_builtin(__builtin_amdgcn_s_wait_asynccnt)
    __builtin_amdgcn_s_wait_asynccnt(0);
#elif defined(HAVE_ASYNC_LDS)
    asm volatile("s_wait_asynccnt 0x0" ::: "memory");
#endif
#endif
}

__device__ __forceinline__ __bf16 f2bf(float f) {
    unsigned u = __builtin_bit_cast(unsigned, f);
    u += 0x7FFFu + ((u >> 16) & 1u);           // round-to-nearest-even
    unsigned short s = (unsigned short)(u >> 16);
    return __builtin_bit_cast(__bf16, s);
}
__device__ __forceinline__ float bf2f(__bf16 b) {
    unsigned short s = __builtin_bit_cast(unsigned short, b);
    unsigned u = ((unsigned)s) << 16;
    return __builtin_bit_cast(float, u);
}

// ---------------------------------------------------------------------------
// f32 -> bf16 conversion (weights / one-time per call)
// ---------------------------------------------------------------------------
__global__ __launch_bounds__(256)
void cvt_bf16_kernel(const float* __restrict__ in, __bf16* __restrict__ out, int n) {
    int i = blockIdx.x * 256 + threadIdx.x;
    if (i < n) out[i] = f2bf(in[i]);
}

// ---------------------------------------------------------------------------
// LayerNorm (row = 1024 floats) -> bf16 output, fused scale/shift
// ---------------------------------------------------------------------------
__global__ __launch_bounds__(256)
void layernorm_bf16_kernel(const float* __restrict__ x,
                           const float* __restrict__ w,
                           const float* __restrict__ b,
                           __bf16* __restrict__ out) {
    __shared__ float red[16];
    const int row = blockIdx.x;
    const float* xr = x + (size_t)row * 1024;
    const int tid = threadIdx.x;
    const int lane = tid & 31, wave = tid >> 5;

    float v[4];
    float s = 0.f;
#pragma unroll
    for (int i = 0; i < 4; i++) { v[i] = xr[tid + 256 * i]; s += v[i]; }
#pragma unroll
    for (int o = 16; o > 0; o >>= 1) s += __shfl_xor(s, o, 32);
    if (lane == 0) red[wave] = s;
    __syncthreads();
    float mu = 0.f;
#pragma unroll
    for (int i = 0; i < 8; i++) mu += red[i];
    mu *= (1.f / 1024.f);

    float s2 = 0.f;
#pragma unroll
    for (int i = 0; i < 4; i++) { float d = v[i] - mu; s2 += d * d; }
#pragma unroll
    for (int o = 16; o > 0; o >>= 1) s2 += __shfl_xor(s2, o, 32);
    if (lane == 0) red[8 + wave] = s2;
    __syncthreads();
    float var = 0.f;
#pragma unroll
    for (int i = 0; i < 8; i++) var += red[8 + i];
    var *= (1.f / 1024.f);
    const float rstd = rsqrtf(var + 1e-5f);

    __bf16* orow = out + (size_t)row * 1024;
#pragma unroll
    for (int i = 0; i < 4; i++) {
        int col = tid + 256 * i;
        orow[col] = f2bf((v[i] - mu) * rstd * w[col] + b[col]);
    }
}

// ---------------------------------------------------------------------------
// Tiled bf16 WMMA GEMM:  C[M,N] = A[M,K] (bf16) x W[N,K]^T (bf16) + bias
// optional exact-erf GELU, optional f32 residual add; f32 and/or bf16 output.
// BM=128 BN=128 BK=32, 256 threads = 8 waves, wave tile 32x64 (2x4 WMMA).
// Double-buffered LDS; tiles streamed with GLOBAL_LOAD_ASYNC_TO_LDS_B128
// (ASYNCcnt) so the DMA overlaps the WMMA pipeline.
// ---------------------------------------------------------------------------
#define BM 128
#define BN 128
#define BK 32
#define LDA 40   // padded LDS K-stride (elements); 80B rows, 16B aligned

__global__ __launch_bounds__(256)
void gemm_bf16_wmma_kernel(const __bf16* __restrict__ A,
                           const __bf16* __restrict__ W,
                           const float*  __restrict__ bias,
                           const float*  __restrict__ residual,
                           float*        __restrict__ outf,
                           __bf16*       __restrict__ outb,
                           int M, int N, int K, int gelu_flag) {
    __shared__ __bf16 As[2][BM][LDA];
    __shared__ __bf16 Bs[2][BN][LDA];

    const int tid  = threadIdx.x;
    const int wave = tid >> 5;
    const int lane = tid & 31;
    const int l15  = lane & 15;
    const int half = lane >> 4;          // 0: lanes 0-15, 1: lanes 16-31
    const int kbase = half * 8;          // per-half K start (ISA 16-bit A/B layout)

    const int wm = (wave >> 1) * 32;     // wave row offset in tile (0/32/64/96)
    const int wn = (wave & 1) * 64;      // wave col offset in tile (0/64)
    const int m0 = blockIdx.y * BM;
    const int n0 = blockIdx.x * BN;

    // this thread's two (row, kchunk) assignments for cooperative staging
    const int r0 = tid >> 2,           kc0 = (tid & 3) * 8;
    const int r1 = (tid + 256) >> 2,   kc1 = (tid & 3) * 8;  // (tid+256)&3 == tid&3

    auto stage = [&](int kk, int bufi) {
        const __bf16* ga0 = A + (size_t)(m0 + r0) * K + kk + kc0;
        const __bf16* gw0 = W + (size_t)(n0 + r0) * K + kk + kc0;
        const __bf16* ga1 = A + (size_t)(m0 + r1) * K + kk + kc1;
        const __bf16* gw1 = W + (size_t)(n0 + r1) * K + kk + kc1;
#if defined(HAVE_ASYNC_LDS)
        __builtin_amdgcn_global_load_async_to_lds_b128(
            (glob128_t)ga0, (lds128_t)&As[bufi][r0][kc0], 0, 0);
        __builtin_amdgcn_global_load_async_to_lds_b128(
            (glob128_t)gw0, (lds128_t)&Bs[bufi][r0][kc0], 0, 0);
        __builtin_amdgcn_global_load_async_to_lds_b128(
            (glob128_t)ga1, (lds128_t)&As[bufi][r1][kc1], 0, 0);
        __builtin_amdgcn_global_load_async_to_lds_b128(
            (glob128_t)gw1, (lds128_t)&Bs[bufi][r1][kc1], 0, 0);
#else
        // fallback: issue all loads before the stores so they overlap
        float4 ta0 = *reinterpret_cast<const float4*>(ga0);
        float4 tw0 = *reinterpret_cast<const float4*>(gw0);
        float4 ta1 = *reinterpret_cast<const float4*>(ga1);
        float4 tw1 = *reinterpret_cast<const float4*>(gw1);
        *reinterpret_cast<float4*>(&As[bufi][r0][kc0]) = ta0;
        *reinterpret_cast<float4*>(&Bs[bufi][r0][kc0]) = tw0;
        *reinterpret_cast<float4*>(&As[bufi][r1][kc1]) = ta1;
        *reinterpret_cast<float4*>(&Bs[bufi][r1][kc1]) = tw1;
#endif
    };

    v8f acc[2][4] = {};

    int buf = 0;
    stage(0, 0);
    wait_async0();
    __syncthreads();

    for (int kk = 0; kk < K; kk += BK) {
        const int next = kk + BK;
        if (next < K) {
            stage(next, buf ^ 1);                       // async DMA next tile
            if (next + BK < K) {                        // warm L2 one tile further
                __builtin_prefetch(A + (size_t)(m0 + r0) * K + next + BK + kc0, 0, 1);
                __builtin_prefetch(W + (size_t)(n0 + r0) * K + next + BK + kc0, 0, 1);
            }
        }

        // fragments: lane L<16 holds K[0..7],K[16..23]; L>=16 holds K[8..15],K[24..31]
        v16bf afrag[2], bfrag[4];
#pragma unroll
        for (int i = 0; i < 2; i++) {
            int r = wm + i * 16 + l15;
            v8bf lo = *reinterpret_cast<const v8bf*>(&As[buf][r][kbase]);
            v8bf hi = *reinterpret_cast<const v8bf*>(&As[buf][r][kbase + 16]);
            afrag[i] = __builtin_shufflevector(lo, hi,
                0,1,2,3,4,5,6,7,8,9,10,11,12,13,14,15);
        }
#pragma unroll
        for (int j = 0; j < 4; j++) {
            int ccol = wn + j * 16 + l15;
            v8bf lo = *reinterpret_cast<const v8bf*>(&Bs[buf][ccol][kbase]);
            v8bf hi = *reinterpret_cast<const v8bf*>(&Bs[buf][ccol][kbase + 16]);
            bfrag[j] = __builtin_shufflevector(lo, hi,
                0,1,2,3,4,5,6,7,8,9,10,11,12,13,14,15);
        }

#pragma unroll
        for (int i = 0; i < 2; i++)
#pragma unroll
            for (int j = 0; j < 4; j++)
                acc[i][j] = __builtin_amdgcn_wmma_f32_16x16x32_bf16(
                    false, afrag[i], false, bfrag[j],
                    (short)0, acc[i][j], false, false);

        if (next < K) {
            wait_async0();          // next tile landed in LDS
            __syncthreads();        // everyone done reading current buffer
            buf ^= 1;
        }
    }

    // epilogue: bias (+GELU) (+residual) -> f32 and/or bf16
#pragma unroll
    for (int i = 0; i < 2; i++) {
#pragma unroll
        for (int j = 0; j < 4; j++) {
            int col = n0 + wn + j * 16 + l15;
            float bc = bias ? bias[col] : 0.f;
#pragma unroll
            for (int r = 0; r < 8; r++) {
                int row = m0 + wm + i * 16 + half * 8 + r;
                float v = acc[i][j][r] + bc;
                if (gelu_flag) v = 0.5f * v * (1.f + erff(v * 0.70710678118f));
                size_t idx = (size_t)row * (size_t)N + col;
                if (residual) v += residual[idx];
                if (outf) outf[idx] = v;
                if (outb) outb[idx] = f2bf(v);
            }
        }
    }
}

// ---------------------------------------------------------------------------
// Local windowed attention (window 128, look-around 1, non-causal).
// qkv: bf16 [B*4096][3072] laid out (3, head, 64). out: bf16 [B*4096][1024].
// grid = (32 windows, 64 batch*head); 128 threads = 1 query each.
// ---------------------------------------------------------------------------
__global__ __launch_bounds__(128)
void local_attn_kernel(const __bf16* __restrict__ qkv, __bf16* __restrict__ out) {
    __shared__ __bf16 Ks[3 * 128][64];
    const int wb = blockIdx.x;             // window block 0..31
    const int bh = blockIdx.y;             // 0..63
    const int b  = bh >> 4, h = bh & 15;
    const int tid = threadIdx.x;
    const int q_tok = wb * 128 + tid;
    const int kg0 = (wb - 1) * 128;

    for (int c = tid; c < 384 * 8; c += 128) {
        int r  = c >> 3;
        int kc = (c & 7) << 3;
        int kg = kg0 + r;
        float4 z = make_float4(0.f, 0.f, 0.f, 0.f);
        if ((unsigned)kg < 4096u)
            z = *reinterpret_cast<const float4*>(
                qkv + (size_t)(b * 4096 + kg) * 3072 + (16 + h) * 64 + kc);
        *reinterpret_cast<float4*>(&Ks[r][kc]) = z;
    }
    __syncthreads();

    float qreg[64];
    {
        const v8bf* qp = reinterpret_cast<const v8bf*>(
            qkv + (size_t)(b * 4096 + q_tok) * 3072 + h * 64);
#pragma unroll
        for (int c = 0; c < 8; c++) {
            v8bf t = qp[c];
#pragma unroll
            for (int d = 0; d < 8; d++) qreg[c * 8 + d] = bf2f(t[d]) * 0.125f;
        }
    }

    const int jlo = (wb == 0)  ? 128 : 0;
    const int jhi = (wb == 31) ? 256 : 384;

    float maxv = -3.0e38f;
    for (int j = jlo; j < jhi; j++) {
        float dot = 0.f;
        const v8bf* kp = reinterpret_cast<const v8bf*>(&Ks[j][0]);
#pragma unroll
        for (int c = 0; c < 8; c++) {
            v8bf kv = kp[c];
#pragma unroll
            for (int d = 0; d < 8; d++) dot += qreg[c * 8 + d] * bf2f(kv[d]);
        }
        maxv = fmaxf(maxv, dot);
    }

    float sum = 0.f;
    float outacc[64];
#pragma unroll
    for (int d = 0; d < 64; d++) outacc[d] = 0.f;

    for (int j = jlo; j < jhi; j++) {
        float dot = 0.f;
        const v8bf* kp = reinterpret_cast<const v8bf*>(&Ks[j][0]);
#pragma unroll
        for (int c = 0; c < 8; c++) {
            v8bf kv = kp[c];
#pragma unroll
            for (int d = 0; d < 8; d++) dot += qreg[c * 8 + d] * bf2f(kv[d]);
        }
        float p = __expf(dot - maxv);
        sum += p;
        const v8bf* vp = reinterpret_cast<const v8bf*>(
            qkv + (size_t)(b * 4096 + kg0 + j) * 3072 + (32 + h) * 64);
#pragma unroll
        for (int c = 0; c < 8; c++) {
            v8bf vv = vp[c];
#pragma unroll
            for (int d = 0; d < 8; d++) outacc[c * 8 + d] += p * bf2f(vv[d]);
        }
    }

    const float inv = 1.f / sum;
    __bf16* orow = out + (size_t)(b * 4096 + q_tok) * 1024 + h * 64;
#pragma unroll
    for (int d = 0; d < 64; d++) orow[d] = f2bf(outacc[d] * inv);
}

// ---------------------------------------------------------------------------
// Host orchestration
// ---------------------------------------------------------------------------
extern "C" void kernel_launch(void* const* d_in, const int* in_sizes, int n_in,
                              void* d_out, int out_size, void* d_ws, size_t ws_size,
                              hipStream_t stream) {
    (void)in_sizes; (void)n_in; (void)out_size; (void)ws_size;

    const float* x      = (const float*)d_in[0];
    const float* ln1_w  = (const float*)d_in[1];
    const float* ln1_b  = (const float*)d_in[2];
    const float* qkv_w  = (const float*)d_in[3];
    const float* qkv_b  = (const float*)d_in[4];
    const float* proj_w = (const float*)d_in[5];
    const float* proj_b = (const float*)d_in[6];
    const float* ln2_w  = (const float*)d_in[7];
    const float* ln2_b  = (const float*)d_in[8];
    const float* fc1_w  = (const float*)d_in[9];
    const float* fc1_b  = (const float*)d_in[10];
    const float* fc2_w  = (const float*)d_in[11];
    const float* fc2_b  = (const float*)d_in[12];
    float* outp = (float*)d_out;

    const int M = 4 * 4096;  // 16384 tokens

    size_t off = 0;
    auto carve = [&](size_t bytes) -> void* {
        void* p = (char*)d_ws + off;
        off += (bytes + 255) & ~(size_t)255;
        return p;
    };
    __bf16* wq   = (__bf16*)carve((size_t)3072 * 1024 * 2);
    __bf16* wp   = (__bf16*)carve((size_t)1024 * 1024 * 2);
    __bf16* w1   = (__bf16*)carve((size_t)4096 * 1024 * 2);
    __bf16* w2   = (__bf16*)carve((size_t)1024 * 4096 * 2);
    __bf16* xn   = (__bf16*)carve((size_t)M * 1024 * 2);
    __bf16* qkvb = (__bf16*)carve((size_t)M * 3072 * 2);
    __bf16* attn = (__bf16*)carve((size_t)M * 1024 * 2);
    float*  xres = (float*) carve((size_t)M * 1024 * 4);
    __bf16* y2   = (__bf16*)carve((size_t)M * 1024 * 2);
    __bf16* hmid = (__bf16*)carve((size_t)M * 4096 * 2);

    {
        int n;
        n = 3072 * 1024; cvt_bf16_kernel<<<(n + 255) / 256, 256, 0, stream>>>(qkv_w,  wq, n);
        n = 1024 * 1024; cvt_bf16_kernel<<<(n + 255) / 256, 256, 0, stream>>>(proj_w, wp, n);
        n = 4096 * 1024; cvt_bf16_kernel<<<(n + 255) / 256, 256, 0, stream>>>(fc1_w,  w1, n);
        n = 1024 * 4096; cvt_bf16_kernel<<<(n + 255) / 256, 256, 0, stream>>>(fc2_w,  w2, n);
    }

    layernorm_bf16_kernel<<<M, 256, 0, stream>>>(x, ln1_w, ln1_b, xn);

    gemm_bf16_wmma_kernel<<<dim3(3072 / BN, M / BM), 256, 0, stream>>>(
        xn, wq, qkv_b, nullptr, nullptr, qkvb, M, 3072, 1024, 0);

    local_attn_kernel<<<dim3(32, 64), 128, 0, stream>>>(qkvb, attn);

    gemm_bf16_wmma_kernel<<<dim3(1024 / BN, M / BM), 256, 0, stream>>>(
        attn, wp, proj_b, x, xres, nullptr, M, 1024, 1024, 0);

    layernorm_bf16_kernel<<<M, 256, 0, stream>>>(xres, ln2_w, ln2_b, y2);

    gemm_bf16_wmma_kernel<<<dim3(4096 / BN, M / BM), 256, 0, stream>>>(
        y2, w1, fc1_b, nullptr, nullptr, hmid, M, 4096, 1024, 1);

    gemm_bf16_wmma_kernel<<<dim3(1024 / BN, M / BM), 256, 0, stream>>>(
        hmid, w2, fc2_b, xres, outp, nullptr, M, 1024, 4096, 0);
}
